// LSTMCell_37099927503216
// MI455X (gfx1250) — compile-verified, matrix operand order checked by reading, not run on hardware
//
#include <hip/hip_runtime.h>
#include <hip/hip_bf16.h>

typedef __bf16 bf16;
typedef __attribute__((ext_vector_type(16))) __bf16 v16bf;
typedef __attribute__((ext_vector_type(8)))  __bf16 v8bf;
typedef __attribute__((ext_vector_type(8)))  float  v8f;
typedef int v4i __attribute__((vector_size(4 * sizeof(int))));   // matches builtin param

#define LDS_AS __attribute__((address_space(3)))
#define GLB_AS __attribute__((address_space(1)))

// Problem sizes (fixed by the reference)
#define BSZ 8192
#define ISZ 1024
#define HSZ 1024
#define KSZ 2048   // I + H after fusing [x | h_prev]
#define MT  256    // workgroup M tile (8 waves x 2 m-subtiles x 16 rows)
#define NT  32     // workgroup N tile (per gate)
#define KB  64     // K elements staged per iteration
#define LDK 72     // padded LDS row length in bf16 (144B -> conflict-spread)

#if __has_builtin(__builtin_amdgcn_global_load_async_to_lds_b128) && \
    __has_builtin(__builtin_amdgcn_s_wait_asynccnt)
#define USE_ASYNC 1
#else
#define USE_ASYNC 0
#endif

// ---- 16B global->LDS copy: async path on CDNA5, sync fallback otherwise
__device__ __forceinline__ void copy16_g2l(bf16* ldst, const bf16* gsrc) {
#if USE_ASYNC
  __builtin_amdgcn_global_load_async_to_lds_b128(
      (GLB_AS v4i*)(unsigned long long)gsrc,
      (LDS_AS v4i*)(unsigned)(unsigned long long)ldst,
      0, 0);
#else
  *(v8bf*)ldst = *(const v8bf*)gsrc;
#endif
}

__device__ __forceinline__ float fast_sigmoid(float z) {
  return 1.0f / (1.0f + __expf(-z));
}
__device__ __forceinline__ float fast_tanh(float z) {
  return 2.0f / (1.0f + __expf(-2.0f * z)) - 1.0f;
}

// ---- pack [x | h_prev] -> bf16 (B, 2048), row-major
__global__ void lstm_pack_xh(const float* __restrict__ x,
                             const float* __restrict__ h,
                             bf16* __restrict__ xh) {
  int c = blockIdx.x * blockDim.x + threadIdx.x;   // one 8-elem chunk
  int idx = c * 8;
  int m = idx >> 11;          // / 2048
  int k = idx & 2047;
  const float* src = (k < ISZ) ? (x + (size_t)m * ISZ + k)
                               : (h + (size_t)m * HSZ + (k - ISZ));
  v8bf o;
#pragma unroll
  for (int j = 0; j < 8; ++j) o[j] = (bf16)src[j];
  *(v8bf*)(xh + idx) = o;
}

// ---- pack weights -> bf16 (4*H, 2048): gate-major rows, [Wx_g | Wh_g] per row
__global__ void lstm_pack_w(const float* __restrict__ Wxi, const float* __restrict__ Whi,
                            const float* __restrict__ Wxf, const float* __restrict__ Whf,
                            const float* __restrict__ Wxc, const float* __restrict__ Whc,
                            const float* __restrict__ Wxo, const float* __restrict__ Who,
                            bf16* __restrict__ wout) {
  int c = blockIdx.x * blockDim.x + threadIdx.x;
  int idx = c * 8;
  int row = idx >> 11;        // 0..4095
  int k   = idx & 2047;
  int g   = row >> 10;        // 0=i,1=f,2=c,3=o
  int n   = row & 1023;
  const float* wx[4] = {Wxi, Wxf, Wxc, Wxo};
  const float* wh[4] = {Whi, Whf, Whc, Who};
  const float* src = (k < ISZ) ? (wx[g] + (size_t)n * ISZ + k)
                               : (wh[g] + (size_t)n * HSZ + (k - ISZ));
  v8bf o;
#pragma unroll
  for (int j = 0; j < 8; ++j) o[j] = (bf16)src[j];
  *(v8bf*)(wout + idx) = o;
}

// ---- fused GEMM (4 gates) + LSTM pointwise epilogue
__launch_bounds__(256)
__global__ void lstm_gemm_fused(const bf16* __restrict__ XH,   // (B, K)
                                const bf16* __restrict__ W,    // (4H, K)
                                const float* __restrict__ c_prev,
                                const float* __restrict__ bxi, const float* __restrict__ bhi,
                                const float* __restrict__ bxf, const float* __restrict__ bhf,
                                const float* __restrict__ bxc, const float* __restrict__ bhc,
                                const float* __restrict__ bxo, const float* __restrict__ bho,
                                float* __restrict__ out) {
  __shared__ bf16 sXH[2][MT * LDK];          // 256 rows of XH tile (36KB/buf)
  __shared__ bf16 sW [2][128 * LDK];         // 4 gates x 32 rows of W tile (18KB/buf)

  const int tid  = threadIdx.x;
  const int wave = tid >> 5;                 // wave32: 8 waves
  const int lane = tid & 31;
  const int mbase = blockIdx.x * MT;
  const int nbase = blockIdx.y * NT;

  // stage KB columns of XH (256 rows) and W (4x32 rows) into LDS buffer `buf`
  auto stage = [&](int buf, int kbase) {
#pragma unroll
    for (int i = 0; i < 8; ++i) {            // XH: 2048 chunks / 256 threads
      int c = tid + i * 256;
      int row = c >> 3, cc = c & 7;
      copy16_g2l(&sXH[buf][row * LDK + cc * 8],
                 XH + (size_t)(mbase + row) * KSZ + kbase + cc * 8);
    }
#pragma unroll
    for (int i = 0; i < 4; ++i) {            // W: 1024 chunks / 256 threads
      int c = tid + i * 256;
      int row = c >> 3, cc = c & 7;          // row: gate = row>>5, r = row&31
      int grow = (row >> 5) * HSZ + nbase + (row & 31);
      copy16_g2l(&sW[buf][row * LDK + cc * 8],
                 W + (size_t)grow * KSZ + kbase + cc * 8);
    }
  };

  v8f acc[2][4][2] = {};                     // [m-subtile][gate][ntile]

  const int half   = lane >> 4;              // 0: lanes 0-15, 1: lanes 16-31
  const int lane16 = lane & 15;
  const int kboff  = half * 8;               // ISA 16-bit A/B fragment K base

  stage(0, 0);
  int buf = 0;
#pragma unroll 1
  for (int kb = 0; kb < KSZ; kb += KB) {
#if USE_ASYNC
    if (kb + KB < KSZ) {
      stage(buf ^ 1, kb + KB);
      __builtin_amdgcn_s_wait_asynccnt(12);  // previous stage's 12 loads landed
    } else {
      __builtin_amdgcn_s_wait_asynccnt(0);
    }
#else
    if (kb + KB < KSZ) stage(buf ^ 1, kb + KB);
#endif
    __syncthreads();

#pragma unroll
    for (int s = 0; s < 2; ++s) {            // two 16x16x32 k-steps per stage
      // A fragments for both m-subtiles (ISA 7.12.2 16-bit A layout:
      // elems 0-7 -> K=kboff+0..7, elems 8-15 -> K=kboff+16..23)
      v16bf a[2];
#pragma unroll
      for (int mt = 0; mt < 2; ++mt) {
        const bf16* arow = &sXH[buf][(wave * 32 + mt * 16 + lane16) * LDK];
        v8bf a0 = *(const v8bf*)(arow + s * 32 + kboff);
        v8bf a1 = *(const v8bf*)(arow + s * 32 + kboff + 16);
        a[mt] = __builtin_shufflevector(a0, a1, 0,1,2,3,4,5,6,7,8,9,10,11,12,13,14,15);
      }
#pragma unroll
      for (int g = 0; g < 4; ++g) {
#pragma unroll
        for (int t = 0; t < 2; ++t) {
          // B fragment (32x16): lane = column n; B = Wᵀ so lane n reads
          // W row (gate, nbase + t*16 + n) contiguously from LDS.
          const bf16* brow = &sW[buf][(g * 32 + t * 16 + lane16) * LDK];
          v8bf b0 = *(const v8bf*)(brow + s * 32 + kboff);
          v8bf b1 = *(const v8bf*)(brow + s * 32 + kboff + 16);
          v16bf b = __builtin_shufflevector(b0, b1, 0,1,2,3,4,5,6,7,8,9,10,11,12,13,14,15);
          acc[0][g][t] = __builtin_amdgcn_wmma_f32_16x16x32_bf16(
              false, a[0], false, b, (short)0, acc[0][g][t], false, false);
          acc[1][g][t] = __builtin_amdgcn_wmma_f32_16x16x32_bf16(
              false, a[1], false, b, (short)0, acc[1][g][t], false, false);
        }
      }
    }
    __syncthreads();
    buf ^= 1;
  }

  // Epilogue: C/D layout — VGPR r, lane l -> row = r + 8*(l>>4), col = l&15
#pragma unroll
  for (int t = 0; t < 2; ++t) {
    int col = nbase + t * 16 + lane16;
    float bi = bxi[col] + bhi[col];
    float bf_ = bxf[col] + bhf[col];
    float bc = bxc[col] + bhc[col];
    float bo = bxo[col] + bho[col];
#pragma unroll
    for (int mt = 0; mt < 2; ++mt) {
#pragma unroll
      for (int r = 0; r < 8; ++r) {
        int row = mbase + wave * 32 + mt * 16 + half * 8 + r;
        float zi = acc[mt][0][t][r] + bi;
        float zf = acc[mt][1][t][r] + bf_;
        float zc = acc[mt][2][t][r] + bc;
        float zo = acc[mt][3][t][r] + bo;
        float ig = fast_sigmoid(zi);
        float fg = fast_sigmoid(zf);
        float gg = fast_tanh(zc);
        float og = fast_sigmoid(zo);
        float cp = c_prev[(size_t)row * HSZ + col];
        float cn = fg * cp + ig * gg;
        float hn = og * fast_tanh(cn);
        out[(size_t)row * HSZ + col] = hn;                          // h_new
        out[(size_t)BSZ * HSZ + (size_t)row * HSZ + col] = cn;      // c_new
      }
    }
  }
}

extern "C" void kernel_launch(void* const* d_in, const int* in_sizes, int n_in,
                              void* d_out, int out_size, void* d_ws, size_t ws_size,
                              hipStream_t stream) {
  const float* x      = (const float*)d_in[0];
  const float* h_prev = (const float*)d_in[1];
  const float* c_prev = (const float*)d_in[2];
  const float* Wxi = (const float*)d_in[3],  *Whi = (const float*)d_in[4];
  const float* Wxf = (const float*)d_in[5],  *Whf = (const float*)d_in[6];
  const float* Wxc = (const float*)d_in[7],  *Whc = (const float*)d_in[8];
  const float* Wxo = (const float*)d_in[9],  *Who = (const float*)d_in[10];
  const float* bxi = (const float*)d_in[11], *bhi = (const float*)d_in[12];
  const float* bxf = (const float*)d_in[13], *bhf = (const float*)d_in[14];
  const float* bxc = (const float*)d_in[15], *bhc = (const float*)d_in[16];
  const float* bxo = (const float*)d_in[17], *bho = (const float*)d_in[18];
  float* out = (float*)d_out;

  // Workspace: XH bf16 (32MB) then packed W bf16 (16MB)
  bf16* XH = (bf16*)d_ws;
  bf16* W  = (bf16*)((char*)d_ws + (size_t)BSZ * KSZ * sizeof(bf16));

  lstm_pack_xh<<<(BSZ * KSZ / 8) / 256, 256, 0, stream>>>(x, h_prev, XH);
  lstm_pack_w<<<(4 * HSZ * KSZ / 8) / 256, 256, 0, stream>>>(
      Wxi, Whi, Wxf, Whf, Wxc, Whc, Wxo, Who, W);

  dim3 grid(BSZ / MT, HSZ / NT);   // 32 x 32 workgroups
  lstm_gemm_fused<<<grid, 256, 0, stream>>>(
      XH, W, c_prev, bxi, bhi, bxf, bhf, bxc, bhc, bxo, bho, out);
}
